// ZipformerBlock_71476845740215
// MI455X (gfx1250) — compile-verified, matrix operand order checked by reading, not run on hardware
//
#include <hip/hip_runtime.h>

// ---------------------------------------------------------------------------
// Zipformer block for MI455X (gfx1250): all matmuls via v_wmma_f32_16x16x32_f16
// ---------------------------------------------------------------------------

typedef __attribute__((ext_vector_type(16))) _Float16 v16h;
typedef __attribute__((ext_vector_type(8)))  _Float16 v8h;
typedef __attribute__((ext_vector_type(8)))  float    v8f;

constexpr int Bc = 8, Tc = 1024, Dc = 512, Hc = 8;
constexpr int QDc = 32, PDc = 4, POSDc = 48, VDc = 64, Kc = 31;
constexpr int FF1c = 1536, FF2c = 2048, FF3c = 2560, NLAc = 384;
constexpr int BT = Bc * Tc;                 // 8192

// ------------------------------ device math --------------------------------

__device__ __forceinline__ float softplusf(float x) {
  return (x > 20.f) ? x : log1pf(__expf(x));
}
__device__ __forceinline__ float swooshL(float x) {
  return softplusf(x - 4.f) - 0.08f * x - 0.035f;
}
__device__ __forceinline__ float swooshR(float x) {
  return softplusf(x - 1.f) - 0.08f * x - 0.313261687f;
}

// ------------------------------ WMMA GEMM ----------------------------------
// C[M,N] = act(alpha * A[M,K] x B[K,N] + bias) + resid   (batched, strided)
// A f32 row-major (lda), B f32 row-major (ldb) or [N,K] if transB,
// C f32 row-major (ldc). Staged into LDS as f16, f32 accumulation.

struct GemmArgs {
  const float* A; const float* B; float* C;
  const float* bias; const float* resid;
  int M, N, K;
  int lda, ldb, ldc;
  long long sAo, sAi, sBo, sBi, sCo, sCi;
  int inner;       // inner batch count (outer*inner = gridDim.z)
  int transB;      // 1: B element (k,n) at B[n*ldb + k]
  int act;         // 0 none, 1 swooshL, 2 swooshR
  float alpha;
};

// ISA 7.12.2 f16 A/B fragment: lane holds row (M or N) = lane&15; halves
// 0..7 = K 8*(lane>>4)..+7 ; halves 8..15 = K 16+8*(lane>>4)..+7.
__device__ __forceinline__ v16h ldfrag(const _Float16* row, int lane) {
  int grp = (lane >> 4) * 8;
  v8h lo = *(const v8h*)(row + grp);
  v8h hi = *(const v8h*)(row + 16 + grp);
  v16h r;
#pragma unroll
  for (int i = 0; i < 8; ++i) { r[i] = lo[i]; r[i + 8] = hi[i]; }
  return r;
}

// Fast staging: dst [ROWS][32] f16 <- src row-major f32 (ld), fully in-bounds.
// Vectorized: global_load_b128 x (32/TPR/4), ds_store_b128.
template <int ROWS>
__device__ __forceinline__ void stage_fast_rowmajor(_Float16* dst,
                                                    const float* src, int ld) {
  constexpr int TPR = 256 / ROWS;        // threads per row (2 or 4)
  constexpr int KPT = 32 / TPR;          // k elems per thread (16 or 8)
  const int tid = threadIdx.x;
  const int row = tid / TPR;
  const int kb  = (tid % TPR) * KPT;
  const float* s = src + (long long)row * ld + kb;
  _Float16*    d = dst + row * 32 + kb;
#pragma unroll
  for (int c = 0; c < KPT; c += 8) {
    float4 f0 = *(const float4*)(s + c);
    float4 f1 = *(const float4*)(s + c + 4);
    v8h h;
    h[0] = (_Float16)f0.x; h[1] = (_Float16)f0.y;
    h[2] = (_Float16)f0.z; h[3] = (_Float16)f0.w;
    h[4] = (_Float16)f1.x; h[5] = (_Float16)f1.y;
    h[6] = (_Float16)f1.z; h[7] = (_Float16)f1.w;
    *(v8h*)(d + c) = h;
  }
}

// Fast staging: dst [ROWSN][32] f16 <- src k-major f32 [32][ld] (element (k,n)
// at src[k*ld+n]), fully in-bounds. Vector global loads along n.
template <int ROWSN>
__device__ __forceinline__ void stage_fast_kmajor(_Float16* dst,
                                                  const float* src, int ld) {
  constexpr int NCH = ROWSN / 8;               // 8-wide n-chunks per k row
  constexpr int KPT = (ROWSN * 32 / 256) / 8;  // k rows per thread (1 or 2)
  const int tid = threadIdx.x;
  const int nc = tid % NCH;
  const int kbase = (tid / NCH) * KPT;
#pragma unroll
  for (int kk = 0; kk < KPT; ++kk) {
    int k = kbase + kk;
    const float* s = src + (long long)k * ld + nc * 8;
    float4 f0 = *(const float4*)(s);
    float4 f1 = *(const float4*)(s + 4);
    _Float16 h[8] = {(_Float16)f0.x, (_Float16)f0.y, (_Float16)f0.z,
                     (_Float16)f0.w, (_Float16)f1.x, (_Float16)f1.y,
                     (_Float16)f1.z, (_Float16)f1.w};
#pragma unroll
    for (int j = 0; j < 8; ++j) dst[(nc * 8 + j) * 32 + k] = h[j];
  }
}

template <int TBM, int TBN, int TWM, int TWN>
__global__ __launch_bounds__(256) void gemm_wmma(GemmArgs g) {
  __shared__ _Float16 lA[TBM * 32];
  __shared__ _Float16 lB[TBN * 32];

  constexpr int MT = TWM / 16;        // A-frags per wave
  constexpr int NT = TWN / 16;        // B-frags per wave
  constexpr int WAVES_N = TBN / TWN;

  const int tid  = threadIdx.x;
  const int lane = tid & 31;
  const int wid  = tid >> 5;          // 8 wave32s
  const int bz   = blockIdx.z;
  const int o    = bz / g.inner;
  const int ii   = bz % g.inner;

  const long long aoff = (long long)o * g.sAo + (long long)ii * g.sAi;
  const long long boff = (long long)o * g.sBo + (long long)ii * g.sBi;
  const long long coff = (long long)o * g.sCo + (long long)ii * g.sCi;

  const int m0 = blockIdx.y * TBM;
  const int n0 = blockIdx.x * TBN;
  const int wm = (wid / WAVES_N) * TWM;
  const int wn = (wid % WAVES_N) * TWN;

  const bool fullM = (m0 + TBM <= g.M);
  const bool fullN = (n0 + TBN <= g.N);

  v8f acc[MT][NT] = {};

  for (int k0 = 0; k0 < g.K; k0 += 32) {
    const bool fullK = (k0 + 32 <= g.K);

    // prefetch next A tile (global_prefetch_b8)
    if (k0 + 32 < g.K) {
      int pr = m0 + (tid >> 1);
      if (pr < g.M)
        __builtin_prefetch(&g.A[aoff + (long long)pr * g.lda + k0 + 32], 0, 1);
    }

    if (fullM && fullK) {
      stage_fast_rowmajor<TBM>(lA, g.A + aoff + (long long)m0 * g.lda + k0,
                               g.lda);
    } else {
      for (int i = tid; i < TBM * 32; i += 256) {
        int r = i >> 5, kk = i & 31;
        int gm = m0 + r, gk = k0 + kk;
        float v = (gm < g.M && gk < g.K)
                    ? g.A[aoff + (long long)gm * g.lda + gk] : 0.f;
        lA[i] = (_Float16)v;
      }
    }

    if (fullN && fullK) {
      if (g.transB)
        stage_fast_rowmajor<TBN>(lB, g.B + boff + (long long)n0 * g.ldb + k0,
                                 g.ldb);
      else
        stage_fast_kmajor<TBN>(lB, g.B + boff + (long long)k0 * g.ldb + n0,
                               g.ldb);
    } else {
      for (int i = tid; i < TBN * 32; i += 256) {
        int n = i >> 5, kk = i & 31;
        int gn = n0 + n, gk = k0 + kk;
        float v = 0.f;
        if (gn < g.N && gk < g.K)
          v = g.transB ? g.B[boff + (long long)gn * g.ldb + gk]
                       : g.B[boff + (long long)gk * g.ldb + gn];
        lB[i] = (_Float16)v;
      }
    }
    __syncthreads();

    v16h af[MT], bf[NT];
#pragma unroll
    for (int i = 0; i < MT; ++i)
      af[i] = ldfrag(&lA[(wm + i * 16 + (lane & 15)) * 32], lane);
#pragma unroll
    for (int j = 0; j < NT; ++j)
      bf[j] = ldfrag(&lB[(wn + j * 16 + (lane & 15)) * 32], lane);

#pragma unroll
    for (int i = 0; i < MT; ++i)
#pragma unroll
      for (int j = 0; j < NT; ++j)
        acc[i][j] = __builtin_amdgcn_wmma_f32_16x16x32_f16(
            false, af[i], false, bf[j], (short)0, acc[i][j], false, false);
    __syncthreads();
  }

  // epilogue: C layout M = v + 8*(lane>>4), N = lane&15 (ISA 7.12.2)
  const int lm = (lane >> 4) * 8;
  const int ln = lane & 15;
  if (fullM && fullN) {
#pragma unroll
    for (int tm = 0; tm < MT; ++tm)
#pragma unroll
      for (int tn = 0; tn < NT; ++tn) {
        int gm0 = m0 + wm + tm * 16 + lm;
        int gn  = n0 + wn + tn * 16 + ln;
        float bv = g.bias ? g.bias[gn] : 0.f;
#pragma unroll
        for (int v = 0; v < 8; ++v) {
          float val = acc[tm][tn][v] * g.alpha + bv;
          if (g.act == 1)      val = swooshL(val);
          else if (g.act == 2) val = swooshR(val);
          long long ci = coff + (long long)(gm0 + v) * g.ldc + gn;
          if (g.resid) val += g.resid[ci];
          g.C[ci] = val;
        }
      }
  } else {
#pragma unroll
    for (int tm = 0; tm < MT; ++tm)
#pragma unroll
      for (int tn = 0; tn < NT; ++tn)
#pragma unroll
        for (int v = 0; v < 8; ++v) {
          int gm = m0 + wm + tm * 16 + lm + v;
          int gn = n0 + wn + tn * 16 + ln;
          if (gm < g.M && gn < g.N) {
            float val = acc[tm][tn][v] * g.alpha;
            if (g.bias) val += g.bias[gn];
            if (g.act == 1)      val = swooshL(val);
            else if (g.act == 2) val = swooshR(val);
            long long ci = coff + (long long)gm * g.ldc + gn;
            if (g.resid) val += g.resid[ci];
            g.C[ci] = val;
          }
        }
  }
}

// --------------------------- elementwise kernels ---------------------------

// scores += p . relg ; apply column mask
__global__ void relscore_kernel(float* sc, const float* p, const float* rel,
                                const unsigned char* mask) {
  long long i = (long long)blockIdx.x * blockDim.x + threadIdx.x;
  const long long total = (long long)Bc * Hc * Tc * Tc;
  if (i >= total) return;
  int s = (int)(i % Tc);
  int t = (int)((i / Tc) % Tc);
  int h = (int)((i / ((long long)Tc * Tc)) % Hc);
  int b = (int)(i / ((long long)Tc * Tc * Hc));
  int j = t - s + (Tc - 1);
  const float* pp = p + ((long long)(b * Tc + t)) * (Hc * PDc) + h * PDc;
  const float* rr = rel + (long long)j * (Hc * PDc) + h * PDc;
  float v = sc[i] + pp[0] * rr[0] + pp[1] * rr[1] + pp[2] * rr[2] + pp[3] * rr[3];
  if (!mask[b * Tc + s]) v = -10000.f;
  sc[i] = v;
}

__global__ __launch_bounds__(256) void softmax_kernel(float* sc,
                                                      const unsigned char* mask) {
  int row = blockIdx.x;                       // [B*H*T)
  int t = row % Tc;
  int b = row / (Hc * Tc);
  float* r = sc + (long long)row * Tc;
  __shared__ float red[256];

  float mx = -1e30f;
  for (int i = threadIdx.x; i < Tc; i += 256) mx = fmaxf(mx, r[i]);
  red[threadIdx.x] = mx; __syncthreads();
  for (int s = 128; s > 0; s >>= 1) {
    if (threadIdx.x < s) red[threadIdx.x] = fmaxf(red[threadIdx.x], red[threadIdx.x + s]);
    __syncthreads();
  }
  mx = red[0]; __syncthreads();

  float sum = 0.f;
  for (int i = threadIdx.x; i < Tc; i += 256) {
    float e = __expf(r[i] - mx); r[i] = e; sum += e;
  }
  red[threadIdx.x] = sum; __syncthreads();
  for (int s = 128; s > 0; s >>= 1) {
    if (threadIdx.x < s) red[threadIdx.x] += red[threadIdx.x + s];
    __syncthreads();
  }
  float scale = (mask[b * Tc + t] ? 1.f : 0.f) / red[0];
  for (int i = threadIdx.x; i < Tc; i += 256) r[i] *= scale;
}

__global__ void mask_kernel(float* x, const unsigned char* mask, int cols, int n) {
  int i = blockIdx.x * blockDim.x + threadIdx.x;
  if (i >= n) return;
  if (!mask[i / cols]) x[i] = 0.f;
}

// GLU: a * sigmoid(g), with time mask
__global__ void glu_kernel(const float* u, float* out, const unsigned char* mask, int n) {
  int i = blockIdx.x * blockDim.x + threadIdx.x;
  if (i >= n) return;
  int row = i / Dc, d = i % Dc;
  float a = u[(long long)row * (2 * Dc) + d];
  float g = u[(long long)row * (2 * Dc) + Dc + d];
  out[i] = mask[row] ? a * (1.f / (1.f + __expf(-g))) : 0.f;
}

// depthwise conv (K=31, pad 15) + bias + swooshR, input/out [B,T,D]
__global__ void dwconv_kernel(const float* a, const float* w, const float* bias,
                              float* out, int n) {
  int i = blockIdx.x * blockDim.x + threadIdx.x;
  if (i >= n) return;
  int row = i / Dc, d = i % Dc;
  int b = row / Tc, t = row % Tc;
  float acc = bias[d];
#pragma unroll
  for (int k = 0; k < Kc; ++k) {
    int tt = t + k - (Kc / 2);
    if ((unsigned)tt < (unsigned)Tc)
      acc += a[((long long)(b * Tc + tt)) * Dc + d] * w[d * Kc + k];
  }
  out[i] = swooshR(acc);
}

// g = tanh(u[:,NLA:2NLA]) * u[:,2NLA:3NLA]
__global__ void tanhmul_kernel(const float* u, float* g, int n) {
  int i = blockIdx.x * blockDim.x + threadIdx.x;
  if (i >= n) return;
  int row = i / NLAc, d = i % NLAc;
  const float* ur = u + (long long)row * (3 * NLAc);
  g[i] = tanhf(ur[NLAc + d]) * ur[2 * NLAc + d];
}

// att0 *= a  (a = u[:, :NLA])
__global__ void amul_kernel(const float* u, float* att0, int n) {
  int i = blockIdx.x * blockDim.x + threadIdx.x;
  if (i >= n) return;
  int row = i / NLAc, d = i % NLAc;
  att0[i] *= u[(long long)row * (3 * NLAc) + d];
}

// x = mask ? res + (x - res)*clip(s, .2, 1) : 0
__global__ void bypass_kernel(const float* res, float* x, const float* scale,
                              const unsigned char* mask, int n) {
  int i = blockIdx.x * blockDim.x + threadIdx.x;
  if (i >= n) return;
  int row = i / Dc, d = i % Dc;
  float s = fminf(fmaxf(scale[d], 0.2f), 1.f);
  float v = res[i] + (x[i] - res[i]) * s;
  x[i] = mask[row] ? v : 0.f;
}

__global__ __launch_bounds__(256) void rmsnorm_kernel(float* x, const float* nb,
                                                      const float* nls) {
  int row = blockIdx.x;
  float* r = x + (long long)row * Dc;
  __shared__ float red[256];
  float s = 0.f;
  for (int i = threadIdx.x; i < Dc; i += 256) {
    float d = r[i] - nb[i]; s += d * d;
  }
  red[threadIdx.x] = s; __syncthreads();
  for (int st = 128; st > 0; st >>= 1) {
    if (threadIdx.x < st) red[threadIdx.x] += red[threadIdx.x + st];
    __syncthreads();
  }
  float sc = __expf(nls[0]) / sqrtf(red[0] / (float)Dc + 1e-8f);
  for (int i = threadIdx.x; i < Dc; i += 256) r[i] *= sc;
}

// ------------------------------- host side ---------------------------------

static inline int cdiv(int a, int b) { return (a + b - 1) / b; }

static void gemm(hipStream_t st, const float* A, const float* B, float* C,
                 const float* bias, const float* resid,
                 int M, int N, int K, int lda, int ldb, int ldc,
                 long long sAo, long long sAi, long long sBo, long long sBi,
                 long long sCo, long long sCi, int outer, int inner,
                 int transB, int act, float alpha) {
  GemmArgs g{A, B, C, bias, resid, M, N, K, lda, ldb, ldc,
             sAo, sAi, sBo, sBi, sCo, sCi, inner, transB, act, alpha};
  if (N % 128 == 0) {
    dim3 grid(N / 128, cdiv(M, 128), outer * inner);
    gemm_wmma<128, 128, 64, 32><<<grid, dim3(256), 0, st>>>(g);
  } else {
    dim3 grid(cdiv(N, 64), cdiv(M, 128), outer * inner);
    gemm_wmma<128, 64, 32, 32><<<grid, dim3(256), 0, st>>>(g);
  }
}

extern "C" void kernel_launch(void* const* d_in, const int* in_sizes, int n_in,
                              void* d_out, int out_size, void* d_ws, size_t ws_size,
                              hipStream_t stream) {
  (void)in_sizes; (void)n_in; (void)out_size; (void)ws_size;
  const float* x   = (const float*)d_in[0];
  const float* pos = (const float*)d_in[1];
  const unsigned char* mask = (const unsigned char*)d_in[2];
  const float* Wq  = (const float*)d_in[3];
  const float* Wk  = (const float*)d_in[4];
  const float* Wpq = (const float*)d_in[5];
  const float* Wpos= (const float*)d_in[6];
  const float* W7  = (const float*)d_in[7];  const float* b8  = (const float*)d_in[8];
  const float* W9  = (const float*)d_in[9];  const float* b10 = (const float*)d_in[10];
  const float* W11 = (const float*)d_in[11]; const float* b12 = (const float*)d_in[12];
  const float* W13 = (const float*)d_in[13]; const float* b14 = (const float*)d_in[14];
  const float* W15 = (const float*)d_in[15]; const float* b16 = (const float*)d_in[16];
  const float* W17 = (const float*)d_in[17]; const float* b18 = (const float*)d_in[18];
  const float* W19 = (const float*)d_in[19]; const float* b20 = (const float*)d_in[20];
  const float* W21 = (const float*)d_in[21]; const float* b22 = (const float*)d_in[22];
  const float* W23 = (const float*)d_in[23]; const float* b24 = (const float*)d_in[24];
  const float* W25 = (const float*)d_in[25]; const float* b26 = (const float*)d_in[26];
  const float* W27 = (const float*)d_in[27]; const float* b28 = (const float*)d_in[28];
  const float* midscale = (const float*)d_in[29];
  const float* W30 = (const float*)d_in[30]; const float* b31 = (const float*)d_in[31];
  const float* W32 = (const float*)d_in[32]; const float* b33 = (const float*)d_in[33];
  const float* W34 = (const float*)d_in[34]; const float* b35 = (const float*)d_in[35];
  const float* W36 = (const float*)d_in[36]; const float* b37 = (const float*)d_in[37];
  const float* W38 = (const float*)d_in[38]; const float* b39 = (const float*)d_in[39];
  const float* W40 = (const float*)d_in[40]; const float* b41 = (const float*)d_in[41];
  const float* W42 = (const float*)d_in[42]; const float* b43 = (const float*)d_in[43];
  const float* normb = (const float*)d_in[44];
  const float* nls   = (const float*)d_in[45];
  const float* outscale = (const float*)d_in[46];

  float* ws = (float*)d_ws;
  size_t off = 0;
  float* attn = ws + off; off += (size_t)Bc * Hc * Tc * Tc;   // 67.1M f32
  float* qb   = ws + off; off += (size_t)BT * (Hc * QDc);
  float* kb   = ws + off; off += (size_t)BT * (Hc * QDc);
  float* pb   = ws + off; off += (size_t)BT * (Hc * PDc);
  float* relb = ws + off; off += (size_t)65536;
  float* t1   = ws + off; off += (size_t)BT * FF3c;           // widest hidden
  float* t2   = ws + off; off += (size_t)BT * Dc;
  float* t3   = ws + off; off += (size_t)BT * Dc;
  float* t4   = ws + off; off += (size_t)BT * Dc;
  float* xcur = (float*)d_out;

  const long long TT = (long long)Tc * Tc;
  const int nBTD = BT * Dc;
  const int EB = 256;

  // ---- attention tensors from the ORIGINAL x ----
  gemm(stream, x, Wq, qb, nullptr, nullptr, BT, Hc * QDc, Dc, Dc, Hc * QDc, Hc * QDc,
       0, 0, 0, 0, 0, 0, 1, 1, 0, 0, 1.f);
  gemm(stream, x, Wk, kb, nullptr, nullptr, BT, Hc * QDc, Dc, Dc, Hc * QDc, Hc * QDc,
       0, 0, 0, 0, 0, 0, 1, 1, 0, 0, 1.f);
  gemm(stream, x, Wpq, pb, nullptr, nullptr, BT, Hc * PDc, Dc, Dc, Hc * PDc, Hc * PDc,
       0, 0, 0, 0, 0, 0, 1, 1, 0, 0, 1.f);
  gemm(stream, pos, Wpos, relb, nullptr, nullptr, 2 * Tc - 1, Hc * PDc, POSDc,
       POSDc, Hc * PDc, Hc * PDc, 0, 0, 0, 0, 0, 0, 1, 1, 0, 0, 1.f);

  // content scores: q . k^T per (b,h), alpha = QD^-0.5
  gemm(stream, qb, kb, attn, nullptr, nullptr, Tc, Tc, QDc,
       Hc * QDc, Hc * QDc, Tc,
       (long long)Tc * Hc * QDc, QDc, (long long)Tc * Hc * QDc, QDc,
       (long long)Hc * TT, TT, Bc, Hc, /*transB=*/1, 0, 0.17677669529663689f);

  {
    long long n = (long long)Bc * Hc * TT;
    relscore_kernel<<<(int)((n + EB - 1) / EB), EB, 0, stream>>>(attn, pb, relb, mask);
  }
  softmax_kernel<<<Bc * Hc * Tc, EB, 0, stream>>>(attn, mask);

  // ---- ff1: x = mk(x + swooshL(x W + b) Wo + bo) ----
  gemm(stream, x, W7, t1, b8, nullptr, BT, FF1c, Dc, Dc, FF1c, FF1c,
       0, 0, 0, 0, 0, 0, 1, 1, 0, 1, 1.f);
  gemm(stream, t1, W9, xcur, b10, x, BT, Dc, FF1c, FF1c, Dc, Dc,
       0, 0, 0, 0, 0, 0, 1, 1, 0, 0, 1.f);
  mask_kernel<<<cdiv(nBTD, EB), EB, 0, stream>>>(xcur, mask, Dc, nBTD);

  // ---- nla ----
  gemm(stream, xcur, W11, t1, b12, nullptr, BT, 3 * NLAc, Dc, Dc, 3 * NLAc, 3 * NLAc,
       0, 0, 0, 0, 0, 0, 1, 1, 0, 0, 1.f);
  {
    int n = BT * NLAc;
    tanhmul_kernel<<<cdiv(n, EB), EB, 0, stream>>>(t1, t3, n);
    // att0[b] = attn[b, h=0] @ g[b]
    gemm(stream, attn, t3, t4, nullptr, nullptr, Tc, NLAc, Tc, Tc, NLAc, NLAc,
         (long long)Hc * TT, 0, (long long)Tc * NLAc, 0, (long long)Tc * NLAc, 0,
         Bc, 1, 0, 0, 1.f);
    amul_kernel<<<cdiv(n, EB), EB, 0, stream>>>(t1, t4, n);
  }
  gemm(stream, t4, W13, xcur, b14, xcur, BT, Dc, NLAc, NLAc, Dc, Dc,
       0, 0, 0, 0, 0, 0, 1, 1, 0, 0, 1.f);
  mask_kernel<<<cdiv(nBTD, EB), EB, 0, stream>>>(xcur, mask, Dc, nBTD);

  // ---- sa1 ----
  gemm(stream, xcur, W15, t1, b16, nullptr, BT, Hc * VDc, Dc, Dc, Hc * VDc, Hc * VDc,
       0, 0, 0, 0, 0, 0, 1, 1, 0, 0, 1.f);
  gemm(stream, attn, t1, t2, nullptr, nullptr, Tc, VDc, Tc, Tc, Hc * VDc, Hc * VDc,
       (long long)Hc * TT, TT, (long long)Tc * Hc * VDc, VDc,
       (long long)Tc * Hc * VDc, VDc, Bc, Hc, 0, 0, 1.f);
  gemm(stream, t2, W17, xcur, b18, xcur, BT, Dc, Hc * VDc, Hc * VDc, Dc, Dc,
       0, 0, 0, 0, 0, 0, 1, 1, 0, 0, 1.f);
  mask_kernel<<<cdiv(nBTD, EB), EB, 0, stream>>>(xcur, mask, Dc, nBTD);

  // ---- conv1 ----
  gemm(stream, xcur, W19, t1, b20, nullptr, BT, 2 * Dc, Dc, Dc, 2 * Dc, 2 * Dc,
       0, 0, 0, 0, 0, 0, 1, 1, 0, 0, 1.f);
  glu_kernel<<<cdiv(nBTD, EB), EB, 0, stream>>>(t1, t2, mask, nBTD);
  dwconv_kernel<<<cdiv(nBTD, EB), EB, 0, stream>>>(t2, W21, b22, t3, nBTD);
  gemm(stream, t3, W23, xcur, b24, xcur, BT, Dc, Dc, Dc, Dc, Dc,
       0, 0, 0, 0, 0, 0, 1, 1, 0, 0, 1.f);
  mask_kernel<<<cdiv(nBTD, EB), EB, 0, stream>>>(xcur, mask, Dc, nBTD);

  // ---- ff2 ----
  gemm(stream, xcur, W25, t1, b26, nullptr, BT, FF2c, Dc, Dc, FF2c, FF2c,
       0, 0, 0, 0, 0, 0, 1, 1, 0, 1, 1.f);
  gemm(stream, t1, W27, xcur, b28, xcur, BT, Dc, FF2c, FF2c, Dc, Dc,
       0, 0, 0, 0, 0, 0, 1, 1, 0, 0, 1.f);
  mask_kernel<<<cdiv(nBTD, EB), EB, 0, stream>>>(xcur, mask, Dc, nBTD);

  // ---- mid bypass ----
  bypass_kernel<<<cdiv(nBTD, EB), EB, 0, stream>>>(x, xcur, midscale, mask, nBTD);

  // ---- sa2 ----
  gemm(stream, xcur, W30, t1, b31, nullptr, BT, Hc * VDc, Dc, Dc, Hc * VDc, Hc * VDc,
       0, 0, 0, 0, 0, 0, 1, 1, 0, 0, 1.f);
  gemm(stream, attn, t1, t2, nullptr, nullptr, Tc, VDc, Tc, Tc, Hc * VDc, Hc * VDc,
       (long long)Hc * TT, TT, (long long)Tc * Hc * VDc, VDc,
       (long long)Tc * Hc * VDc, VDc, Bc, Hc, 0, 0, 1.f);
  gemm(stream, t2, W32, xcur, b33, xcur, BT, Dc, Hc * VDc, Hc * VDc, Dc, Dc,
       0, 0, 0, 0, 0, 0, 1, 1, 0, 0, 1.f);
  mask_kernel<<<cdiv(nBTD, EB), EB, 0, stream>>>(xcur, mask, Dc, nBTD);

  // ---- conv2 ----
  gemm(stream, xcur, W34, t1, b35, nullptr, BT, 2 * Dc, Dc, Dc, 2 * Dc, 2 * Dc,
       0, 0, 0, 0, 0, 0, 1, 1, 0, 0, 1.f);
  glu_kernel<<<cdiv(nBTD, EB), EB, 0, stream>>>(t1, t2, mask, nBTD);
  dwconv_kernel<<<cdiv(nBTD, EB), EB, 0, stream>>>(t2, W36, b37, t3, nBTD);
  gemm(stream, t3, W38, xcur, b39, xcur, BT, Dc, Dc, Dc, Dc, Dc,
       0, 0, 0, 0, 0, 0, 1, 1, 0, 0, 1.f);
  mask_kernel<<<cdiv(nBTD, EB), EB, 0, stream>>>(xcur, mask, Dc, nBTD);

  // ---- ff3 ----
  gemm(stream, xcur, W40, t1, b41, nullptr, BT, FF3c, Dc, Dc, FF3c, FF3c,
       0, 0, 0, 0, 0, 0, 1, 1, 0, 1, 1.f);
  gemm(stream, t1, W42, xcur, b43, xcur, BT, Dc, FF3c, FF3c, Dc, Dc,
       0, 0, 0, 0, 0, 0, 1, 1, 0, 0, 1.f);
  mask_kernel<<<cdiv(nBTD, EB), EB, 0, stream>>>(xcur, mask, Dc, nBTD);

  // ---- norm + out bypass ----
  rmsnorm_kernel<<<BT, EB, 0, stream>>>(xcur, normb, nls);
  bypass_kernel<<<cdiv(nBTD, EB), EB, 0, stream>>>(x, xcur, outscale, mask, nBTD);
}
